// MultiHeadLinearAttention_62302795596629
// MI455X (gfx1250) — matrix operand order repaired and checked by reading, not compiled, not run on hardware
//
#include <hip/hip_runtime.h>

// ---------------------------------------------------------------------------
// MI455X (gfx1250) Performer causal linear attention, bf16 WMMA everywhere.
// Problem: B=1, T=2048, D=1024, H=16, DK=64, M=64. Chunked scan, chunk C=64.
// TDM (tensor_load_to_lds) stages attention chunk tiles into LDS.
// ---------------------------------------------------------------------------

typedef __bf16 bf16_t;
typedef __attribute__((ext_vector_type(16))) __bf16 v16bf;
typedef __attribute__((ext_vector_type(8)))  __bf16 v8bf;
typedef __attribute__((ext_vector_type(8)))  float  v8f;
typedef __attribute__((ext_vector_type(4)))  unsigned int u32x4;
typedef __attribute__((ext_vector_type(8)))  int i32x8;
typedef __attribute__((ext_vector_type(4)))  int i32x4;

#define DEV __device__ __forceinline__

static constexpr int T_SEQ = 2048;
static constexpr int DMODEL = 1024;
static constexpr int NHEAD = 16;
static constexpr int DK = 64;
static constexpr int MFEAT = 64;
static constexpr int CHUNK = 64;

DEV v8f wmma_bf16(v16bf a, v16bf b, v8f c) {
  return __builtin_amdgcn_wmma_f32_16x16x32_bf16(
      /*neg_a=*/false, a, /*neg_b=*/false, b,
      /*c_mod=*/(short)0, c, /*reuse_a=*/false, /*reuse_b=*/false);
}

// A fragment (16x32 bf16): lane L holds row L%16; k = ab..ab+15 contiguous
DEV v16bf a_frag(const bf16_t* p, int ld, int row, int k0, int lane) {
  int ab = (lane < 16) ? 0 : 16;
  const bf16_t* base = p + (size_t)row * ld + k0 + ab;
  v8bf lo = *(const v8bf*)(base);
  v8bf hi = *(const v8bf*)(base + 8);
  v16bf a;
#pragma unroll
  for (int e = 0; e < 8; ++e) { a[e] = lo[e]; a[8 + e] = hi[e]; }
  return a;
}

// B fragment where source is row-major and B = src^T (lane col n = src row),
// so the 16 consecutive k values are contiguous in memory.
DEV v16bf b_frag_rows(const bf16_t* p, int ld, int row, int k0, int lane) {
  int bb = (lane < 16) ? 0 : 16;
  const bf16_t* base = p + (size_t)row * ld + k0 + bb;
  v8bf lo = *(const v8bf*)(base);
  v8bf hi = *(const v8bf*)(base + 8);
  v16bf b;
#pragma unroll
  for (int e = 0; e < 8; ++e) { b[e] = lo[e]; b[8 + e] = hi[e]; }
  return b;
}

// B fragment where the source *is* B row-major ([k][n]): strided per-lane column.
DEV v16bf b_frag_strided(const bf16_t* p, int ld, int n, int k0, int lane) {
  int bb = (lane < 16) ? 0 : 16;
  v16bf b;
#pragma unroll
  for (int e = 0; e < 16; ++e) b[e] = p[(size_t)(k0 + bb + e) * ld + n];
  return b;
}

DEV v16bf b_frag_strided_f32(const float* p, int ld, int n, int k0, int lane) {
  int bb = (lane < 16) ? 0 : 16;
  v16bf b;
#pragma unroll
  for (int e = 0; e < 16; ++e) b[e] = (bf16_t)p[(size_t)(k0 + bb + e) * ld + n];
  return b;
}

// A fragment of src^T where src is [k][m] row-major: a[m][k] = src[k*ld + m].
DEV v16bf aT_frag(const bf16_t* p, int ld, int m, int k0, int lane) {
  int ab = (lane < 16) ? 0 : 16;
  v16bf a;
#pragma unroll
  for (int e = 0; e < 8; ++e) {
    a[e]     = p[(size_t)(k0 + ab + e) * ld + m];
    a[8 + e] = p[(size_t)(k0 + ab + 8 + e) * ld + m];
  }
  return a;
}

DEV void store_tile(float* C, int ldc, int m0, int n0, v8f acc, int lane) {
  int n = n0 + (lane & 15);
  int mb = m0 + ((lane < 16) ? 0 : 8);
#pragma unroll
  for (int i = 0; i < 8; ++i) C[(size_t)(mb + i) * ldc + n] = acc[i];
}

// ---------------------------------------------------------------------------
// Tensor Data Mover: 2-D tile load Global -> LDS via D# descriptor.
// Group0: count=1 | lds_addr | global_addr(57b) | type=2.
// Group1: data_size, tensor_dim0/1, tile_dim0/1, tensor_dim0_stride.
// tile_d1==0 -> 1-D transfer. All dims/strides in data_size units.
// ---------------------------------------------------------------------------
#if defined(__has_builtin)
#if __has_builtin(__builtin_amdgcn_tensor_load_to_lds)
#define HAVE_TDM 1
#endif
#endif

DEV void tdm_load_to_lds(const void* gsrc, void* lds_dst,
                         unsigned tile_d0, unsigned tile_d1,
                         unsigned tensor_d0, unsigned tensor_d1,
                         unsigned stride0_units, unsigned data_size_code) {
#ifdef HAVE_TDM
  unsigned long long ga = (unsigned long long)(uintptr_t)gsrc;
  unsigned ldsoff = (unsigned)(uintptr_t)lds_dst;  // LDS aperture: low 32 bits
  u32x4 g0;
  g0[0] = 1u;                                       // count=1 (valid), user mode
  g0[1] = ldsoff;                                   // lds_addr
  g0[2] = (unsigned)ga;                             // global_addr[31:0]
  g0[3] = (unsigned)((ga >> 32) & 0x01FFFFFFu) | (2u << 30);  // addr[56:32]|type=2
  i32x8 g1;
  g1[0] = (int)(data_size_code << 16);              // wg_mask=0, data_size
  g1[1] = (int)((tensor_d0 & 0xFFFFu) << 16);       // tensor_dim0[15:0] @ [63:48]
  g1[2] = (int)(((tensor_d0 >> 16) & 0xFFFFu) | ((tensor_d1 & 0xFFFFu) << 16));
  g1[3] = (int)(((tensor_d1 >> 16) & 0xFFFFu) | ((tile_d0 & 0xFFFFu) << 16));
  g1[4] = (int)(tile_d1 & 0xFFFFu);                 // tile_dim1 | tile_dim2=0
  g1[5] = (int)stride0_units;                       // tensor_dim0_stride[31:0]
  g1[6] = 0;                                        // stride0[47:32] | stride1[15:0]
  g1[7] = 0;
  i32x4 z4 = {0, 0, 0, 0};
#if defined(__clang_major__) && __clang_major__ >= 23
  i32x8 z8 = {0, 0, 0, 0, 0, 0, 0, 0};
  __builtin_amdgcn_tensor_load_to_lds(g0, g1, z4, z4, z8, 0);
#else
  __builtin_amdgcn_tensor_load_to_lds(g0, g1, z4, z4, 0);
#endif
#else
  (void)gsrc; (void)lds_dst; (void)tile_d0; (void)tile_d1;
  (void)tensor_d0; (void)tensor_d1; (void)stride0_units; (void)data_size_code;
#endif
}

// ---------------------------------------------------------------------------
// Kernel: f32 -> bf16 elementwise convert (grid-stride).
// ---------------------------------------------------------------------------
__global__ void k_cvt_bf16(const float* __restrict__ src,
                           bf16_t* __restrict__ dst, int n) {
  int i = blockIdx.x * blockDim.x + threadIdx.x;
  int stride = gridDim.x * blockDim.x;
  for (; i < n; i += stride) dst[i] = (bf16_t)src[i];
}

// ---------------------------------------------------------------------------
// Kernel: 2048x1024x1024 GEMM, C = A * W^T, bf16 inputs, f32 out.
// 32x32 tile per wave (2x2 WMMA tiles): 4 wmma per 8 b128 loads.
// Block = 8 waves (2x4) covering 64(M) x 128(N).
// ---------------------------------------------------------------------------
__global__ __launch_bounds__(256) void k_gemm2048(
    const bf16_t* __restrict__ A, const bf16_t* __restrict__ W,
    float* __restrict__ C) {
  int lane = threadIdx.x & 31;
  int wave = threadIdx.x >> 5;
  int wr = wave >> 2, wc = wave & 3;
  int t0 = blockIdx.x * 64 + wr * 32;
  int n0 = blockIdx.y * 128 + wc * 32;
  int ar0 = t0 + (lane & 15), ar1 = ar0 + 16;
  int br0 = n0 + (lane & 15), br1 = br0 + 16;
  v8f acc00 = {0.f, 0.f, 0.f, 0.f, 0.f, 0.f, 0.f, 0.f};
  v8f acc01 = acc00, acc10 = acc00, acc11 = acc00;
  for (int kk = 0; kk < DMODEL; kk += 32) {
    __builtin_prefetch(A + (size_t)ar0 * DMODEL + kk + 128, 0, 0);
    __builtin_prefetch(W + (size_t)br0 * DMODEL + kk + 128, 0, 0);
    v16bf a0 = a_frag(A, DMODEL, ar0, kk, lane);
    v16bf a1 = a_frag(A, DMODEL, ar1, kk, lane);
    v16bf b0 = b_frag_rows(W, DMODEL, br0, kk, lane);
    v16bf b1 = b_frag_rows(W, DMODEL, br1, kk, lane);
    acc00 = wmma_bf16(a0, b0, acc00);
    acc01 = wmma_bf16(a0, b1, acc01);
    acc10 = wmma_bf16(a1, b0, acc10);
    acc11 = wmma_bf16(a1, b1, acc11);
  }
  store_tile(C, DMODEL, t0,      n0,      acc00, lane);
  store_tile(C, DMODEL, t0,      n0 + 16, acc01, lane);
  store_tile(C, DMODEL, t0 + 16, n0,      acc10, lane);
  store_tile(C, DMODEL, t0 + 16, n0 + 16, acc11, lane);
}

// ---------------------------------------------------------------------------
// Kernel: per-(t,h) row L2 normalize of 64-wide head slices.
// One wave per row (32 lanes x 2 elems); writes bf16 normalized + nsq.
// ---------------------------------------------------------------------------
__global__ void k_normalize(const float* __restrict__ src,
                            bf16_t* __restrict__ dst,
                            float* __restrict__ nsq, int rows) {
  int wv = (blockIdx.x * blockDim.x + threadIdx.x) >> 5;
  int lane = threadIdx.x & 31;
  if (wv >= rows) return;
  int t = wv >> 4, h = wv & 15;  // rows = T*H, row = t*16 + h
  const float* p = src + (size_t)t * DMODEL + h * DK;
  float2 v = ((const float2*)p)[lane];
  float s = v.x * v.x + v.y * v.y;
#pragma unroll
  for (int off = 16; off >= 1; off >>= 1) s += __shfl_xor(s, off, 32);
  float inv = 1.0f / fmaxf(sqrtf(s), 1e-12f);
  bf16_t* q = dst + (size_t)t * DMODEL + h * DK;
  q[lane * 2]     = (bf16_t)(v.x * inv);
  q[lane * 2 + 1] = (bf16_t)(v.y * inv);
  if (lane == 0) nsq[wv] = s * inv * inv;
}

// ---------------------------------------------------------------------------
// Kernel: FAVOR feature map  out[h][t][m] = exp(qn . omega[h][m] - 0.5*nsq)/8
// grid (T/16, 1, H); 4 waves cover M=64.
// ---------------------------------------------------------------------------
__global__ __launch_bounds__(128) void k_feature(
    const bf16_t* __restrict__ qn, const bf16_t* __restrict__ omg,
    const float* __restrict__ nsq, bf16_t* __restrict__ out) {
  int lane = threadIdx.x & 31;
  int wave = threadIdx.x >> 5;  // 0..3 -> feature-tile
  int t0 = blockIdx.x * 16;
  int h = blockIdx.z;
  int n0 = wave * 16;
  int arow = t0 + (lane & 15);
  int brow = n0 + (lane & 15);
  const bf16_t* om = omg + (size_t)h * MFEAT * DK;
  v8f acc = {0.f, 0.f, 0.f, 0.f, 0.f, 0.f, 0.f, 0.f};
#pragma unroll
  for (int kk = 0; kk < DK; kk += 32) {
    v16bf a = a_frag(qn, DMODEL, arow, h * DK + kk, lane);
    v16bf b = b_frag_rows(om, DK, brow, kk, lane);
    acc = wmma_bf16(a, b, acc);
  }
  int n = n0 + (lane & 15);
  int mb = (lane < 16) ? 0 : 8;
  bf16_t* dst = out + (size_t)h * T_SEQ * MFEAT;
#pragma unroll
  for (int i = 0; i < 8; ++i) {
    int t = t0 + mb + i;
    float val = __expf(acc[i] - 0.5f * nsq[(size_t)t * NHEAD + h]) * 0.125f;
    dst[(size_t)t * MFEAT + n] = (bf16_t)val;
  }
}

// ---------------------------------------------------------------------------
// Kernel: chunked causal linear attention. One block per head, 8 waves.
// Per chunk c: A = tril(Qp Kp^T); den = rowsum(A) + Qp.z;
//   out = (A V + Qp S) / den;  S += Kp^T V;  z += colsum(Kp).
// Chunk tiles staged into LDS via the Tensor Data Mover.
// ---------------------------------------------------------------------------
__global__ __launch_bounds__(256) void k_attn(
    const bf16_t* __restrict__ Qp, const bf16_t* __restrict__ Kp,
    const bf16_t* __restrict__ Vb, float* __restrict__ Out) {
  __shared__ bf16_t sK[CHUNK * CHUNK];
  __shared__ bf16_t sV[CHUNK * CHUNK];
  __shared__ bf16_t sA[CHUNK * CHUNK];
  __shared__ float  sS[MFEAT * DK];
  __shared__ float  sz[MFEAT];
  __shared__ float  sden[CHUNK];

  int h = blockIdx.x;
  int tid = threadIdx.x;
  int lane = tid & 31;
  int wave = tid >> 5;
  const bf16_t* Qh = Qp + (size_t)h * T_SEQ * MFEAT;
  const bf16_t* Kh = Kp + (size_t)h * T_SEQ * MFEAT;

  for (int i = tid; i < MFEAT * DK; i += 256) sS[i] = 0.0f;
  if (tid < MFEAT) sz[tid] = 0.0f;
  __syncthreads();

  for (int c = 0; c < T_SEQ / CHUNK; ++c) {
    int tbase = c * CHUNK;

#ifdef HAVE_TDM
    // TDM stage: Kp chunk is one contiguous 8 KB block (1-D, 8-byte units);
    // V chunk is 64 rows x 128 B with a 2 KB global row stride (2-D).
    if (wave == 0) {
      tdm_load_to_lds(Kh + (size_t)tbase * MFEAT, sK,
                      /*tile_d0=*/1024, /*tile_d1=*/0,
                      /*tensor_d0=*/1024, /*tensor_d1=*/1,
                      /*stride0=*/1024, /*data_size=8B*/ 3);
      tdm_load_to_lds(Vb + (size_t)tbase * DMODEL + h * DK, sV,
                      /*tile_d0=*/16, /*tile_d1=*/64,
                      /*tensor_d0=*/16, /*tensor_d1=*/64,
                      /*stride0=*/256, /*data_size=8B*/ 3);
      __builtin_amdgcn_s_wait_tensorcnt(0);
    }
#else
    {  // cooperative fallback copy (4 threads/row, 32 B each)
      int r = tid >> 2;
      int cp = (tid & 3) * 16;
      const v8bf* srcK = (const v8bf*)(Kh + (size_t)(tbase + r) * MFEAT + cp);
      v8bf* dK = (v8bf*)(sK + r * CHUNK + cp);
      dK[0] = srcK[0]; dK[1] = srcK[1];
      const v8bf* srcV =
          (const v8bf*)(Vb + (size_t)(tbase + r) * DMODEL + h * DK + cp);
      v8bf* dV = (v8bf*)(sV + r * CHUNK + cp);
      dV[0] = srcV[0]; dV[1] = srcV[1];
    }
#endif
    __syncthreads();

    // Phase 1: A = tril(Qp_c Kp_c^T), store bf16 into sA
    for (int tile = wave * 2; tile < wave * 2 + 2; ++tile) {
      int m0 = (tile >> 2) * 16, n0 = (tile & 3) * 16;
      int arow = tbase + m0 + (lane & 15);
      v8f acc = {0.f, 0.f, 0.f, 0.f, 0.f, 0.f, 0.f, 0.f};
#pragma unroll
      for (int kk = 0; kk < MFEAT; kk += 32) {
        v16bf a = a_frag(Qh, MFEAT, arow, kk, lane);
        v16bf b = b_frag_rows(sK, CHUNK, n0 + (lane & 15), kk, lane);
        acc = wmma_bf16(a, b, acc);
      }
      int n = n0 + (lane & 15);
      int mb = m0 + ((lane < 16) ? 0 : 8);
#pragma unroll
      for (int i = 0; i < 8; ++i) {
        int m = mb + i;
        sA[m * CHUNK + n] = (bf16_t)((n <= m) ? acc[i] : 0.0f);  // causal mask
      }
    }
    __syncthreads();

    // den_t = rowsum(sA) + Qp_t . z   (uses z BEFORE this chunk's update)
    if (tid < CHUNK) {
      int t = tid;
      float rs = 0.f;
      for (int s = 0; s < CHUNK; ++s) rs += (float)sA[t * CHUNK + s];
      const bf16_t* qrow = Qh + (size_t)(tbase + t) * MFEAT;
      float qz = 0.f;
      for (int m = 0; m < MFEAT; ++m) qz += (float)qrow[m] * sz[m];
      sden[t] = fmaxf(rs + qz, 1e-6f);
    }
    __syncthreads();

    // Phase 2: out = (sA * V + Qp * S) / den
    for (int tile = wave * 2; tile < wave * 2 + 2; ++tile) {
      int m0 = (tile >> 2) * 16, n0 = (tile & 3) * 16;
      int lrow = m0 + (lane & 15);
      int n = n0 + (lane & 15);
      v8f acc = {0.f, 0.f, 0.f, 0.f, 0.f, 0.f, 0.f, 0.f};
#pragma unroll
      for (int kk = 0; kk < CHUNK; kk += 32) {
        v16bf a = a_frag(sA, CHUNK, lrow, kk, lane);
        v16bf b = b_frag_strided(sV, DK, n, kk, lane);
        acc = wmma_bf16(a, b, acc);
      }
#pragma unroll
      for (int kk = 0; kk < MFEAT; kk += 32) {
        v16bf a = a_frag(Qh, MFEAT, tbase + lrow, kk, lane);
        v16bf b = b_frag_strided_f32(sS, DK, n, kk, lane);
        acc = wmma_bf16(a, b, acc);
      }
      int mb = m0 + ((lane < 16) ? 0 : 8);
#pragma unroll
      for (int i = 0; i < 8; ++i) {
        int m = mb + i;
        Out[(size_t)(tbase + m) * DMODEL + h * DK + n] = acc[i] / sden[m];
      }
    }
    __syncthreads();

    // Phase 3: S += Kp^T V ; z += colsum(Kp)
    for (int tile = wave * 2; tile < wave * 2 + 2; ++tile) {
      int m0 = (tile >> 2) * 16, n0 = (tile & 3) * 16;
      int n = n0 + (lane & 15);
      int mb = m0 + ((lane < 16) ? 0 : 8);
      v8f acc;
#pragma unroll
      for (int i = 0; i < 8; ++i) acc[i] = sS[(mb + i) * DK + n];
      int mrow = m0 + (lane & 15);
#pragma unroll
      for (int kk = 0; kk < CHUNK; kk += 32) {
        v16bf a = aT_frag(sK, CHUNK, mrow, kk, lane);
        v16bf b = b_frag_strided(sV, DK, n, kk, lane);
        acc = wmma_bf16(a, b, acc);
      }
#pragma unroll
      for (int i = 0; i < 8; ++i) sS[(mb + i) * DK + n] = acc[i];
    }
    if (tid < MFEAT) {
      int m = tid;
      float s = 0.f;
      for (int t = 0; t < CHUNK; ++t) s += (float)sK[t * CHUNK + m];
      sz[m] += s;
    }
    __syncthreads();
  }
}

// ---------------------------------------------------------------------------
// Host driver
// ---------------------------------------------------------------------------
extern "C" void kernel_launch(void* const* d_in, const int* in_sizes, int n_in,
                              void* d_out, int out_size, void* d_ws,
                              size_t ws_size, hipStream_t stream) {
  (void)in_sizes; (void)n_in; (void)out_size; (void)ws_size;
  const float* x   = (const float*)d_in[0];
  const float* Wq  = (const float*)d_in[1];
  const float* Wk  = (const float*)d_in[2];
  const float* Wv  = (const float*)d_in[3];
  const float* Wo  = (const float*)d_in[4];
  const float* omg = (const float*)d_in[5];
  float* out = (float*)d_out;

  char* ws = (char*)d_ws;
  auto alloc = [&](size_t bytes) -> char* {
    char* p = ws;
    ws += (bytes + 255) & ~(size_t)255;
    return p;
  };
  const size_t TD = (size_t)T_SEQ * DMODEL;
  const size_t DD = (size_t)DMODEL * DMODEL;
  const size_t HTM = (size_t)NHEAD * T_SEQ * MFEAT;

  bf16_t* Xb  = (bf16_t*)alloc(TD * 2);
  bf16_t* Wqb = (bf16_t*)alloc(DD * 2);
  bf16_t* Wkb = (bf16_t*)alloc(DD * 2);
  bf16_t* Wvb = (bf16_t*)alloc(DD * 2);
  bf16_t* Wob = (bf16_t*)alloc(DD * 2);
  bf16_t* Ob  = (bf16_t*)alloc((size_t)NHEAD * MFEAT * DK * 2);
  float*  Qf  = (float*)alloc(TD * 4);
  float*  Kf  = (float*)alloc(TD * 4);
  float*  Vf  = (float*)alloc(TD * 4);
  bf16_t* Qnb = (bf16_t*)alloc(TD * 2);
  bf16_t* Knb = (bf16_t*)alloc(TD * 2);
  bf16_t* Vb  = (bf16_t*)alloc(TD * 2);
  float*  nsqQ = (float*)alloc((size_t)T_SEQ * NHEAD * 4);
  float*  nsqK = (float*)alloc((size_t)T_SEQ * NHEAD * 4);
  bf16_t* QpB = (bf16_t*)alloc(HTM * 2);
  bf16_t* KpB = (bf16_t*)alloc(HTM * 2);
  float*  AttF = (float*)alloc(TD * 4);
  bf16_t* AttB = (bf16_t*)alloc(TD * 2);

  // 1) bf16 conversions of inputs
  k_cvt_bf16<<<2048, 256, 0, stream>>>(x, Xb, (int)TD);
  k_cvt_bf16<<<2048, 256, 0, stream>>>(Wq, Wqb, (int)DD);
  k_cvt_bf16<<<2048, 256, 0, stream>>>(Wk, Wkb, (int)DD);
  k_cvt_bf16<<<2048, 256, 0, stream>>>(Wv, Wvb, (int)DD);
  k_cvt_bf16<<<2048, 256, 0, stream>>>(Wo, Wob, (int)DD);
  k_cvt_bf16<<<256, 256, 0, stream>>>(omg, Ob, NHEAD * MFEAT * DK);

  // 2) Q/K/V projections (WMMA GEMMs), 32x32 tile per wave
  dim3 ggrid(T_SEQ / 64, DMODEL / 128);
  k_gemm2048<<<ggrid, 256, 0, stream>>>(Xb, Wqb, Qf);
  k_gemm2048<<<ggrid, 256, 0, stream>>>(Xb, Wkb, Kf);
  k_gemm2048<<<ggrid, 256, 0, stream>>>(Xb, Wvb, Vf);

  // 3) L2-normalize q,k per head-row; convert v to bf16
  int rows = T_SEQ * NHEAD;
  k_normalize<<<rows / 8, 256, 0, stream>>>(Qf, Qnb, nsqQ, rows);
  k_normalize<<<rows / 8, 256, 0, stream>>>(Kf, Knb, nsqK, rows);
  k_cvt_bf16<<<2048, 256, 0, stream>>>(Vf, Vb, (int)TD);

  // 4) FAVOR feature maps (WMMA)
  dim3 fgrid(T_SEQ / 16, 1, NHEAD);
  k_feature<<<fgrid, 128, 0, stream>>>(Qnb, Ob, nsqQ, QpB);
  k_feature<<<fgrid, 128, 0, stream>>>(Knb, Ob, nsqK, KpB);

  // 5) chunked causal linear attention (WMMA + TDM), one block per head
  k_attn<<<NHEAD, 256, 0, stream>>>(QpB, KpB, Vb, AttF);

  // 6) output projection (WMMA GEMM) -> d_out (f32)
  k_cvt_bf16<<<2048, 256, 0, stream>>>(AttF, AttB, (int)TD);
  k_gemm2048<<<ggrid, 256, 0, stream>>>(AttB, Wob, out);
}
